// SingleStreamBlock_58729382805620
// MI455X (gfx1250) — matrix operand order repaired and checked by reading, not compiled, not run on hardware
//
#include <hip/hip_runtime.h>
#include <hip/hip_bf16.h>
#include <math.h>

typedef __bf16 bf16;
typedef __attribute__((ext_vector_type(16))) __bf16 v16bf;
typedef __attribute__((ext_vector_type(8)))  __bf16 v8bf;
typedef __attribute__((ext_vector_type(8)))  float   v8f;

union V16U { v16bf v; v8bf h[2]; };
static __device__ __forceinline__ v16bf combine16(v8bf lo, v8bf hi){
    V16U u; u.h[0] = lo; u.h[1] = hi; return u.v;
}

// async copy 16B global -> LDS (gfx1250 GLOBAL_LOAD_ASYNC_TO_LDS_B128, ASYNCcnt)
static __device__ __forceinline__ void async_g2l_b128(unsigned lds_byte_off,
                                                      const void* gaddr){
    unsigned long long ga = (unsigned long long)(size_t)gaddr;
    asm volatile("global_load_async_to_lds_b128 %0, %1, off"
                 :: "v"(lds_byte_off), "v"(ga) : "memory");
}
static __device__ __forceinline__ void wait_async0(){
    asm volatile("s_wait_asynccnt 0x0" ::: "memory");
}

// ---------------------------------------------------------------------------
// f32 -> bf16 elementwise convert (weights)
// ---------------------------------------------------------------------------
__global__ __launch_bounds__(256)
void f2bf_kernel(const float* __restrict__ in, bf16* __restrict__ out, size_t n){
    size_t i = (size_t)blockIdx.x * blockDim.x + threadIdx.x;
    size_t stride = (size_t)gridDim.x * blockDim.x;
    for (; i < n; i += stride) out[i] = (bf16)in[i];
}

// ---------------------------------------------------------------------------
// mod = silu(vec) @ mod_w.T + mod_b   (8 x 3072)
// ---------------------------------------------------------------------------
__global__ __launch_bounds__(256)
void mod_kernel(const float* __restrict__ vec, const float* __restrict__ mod_w,
                const float* __restrict__ mod_b, float* __restrict__ mod){
    __shared__ float sv[1024];
    const int b = blockIdx.y;
    for (int i = threadIdx.x; i < 1024; i += 256){
        float v = vec[b*1024 + i];
        sv[i] = v / (1.0f + __expf(-v));
    }
    __syncthreads();
    const int j = blockIdx.x*256 + threadIdx.x;        // 0..3071
    const float* wrow = mod_w + (size_t)j*1024;
    float acc = mod_b[j];
    for (int i = 0; i < 1024; ++i) acc += sv[i]*wrow[i];
    mod[b*3072 + j] = acc;
}

// ---------------------------------------------------------------------------
// modulated LayerNorm: xm = (1+scale)*norm(x) + shift   (bf16 out)
// one block per row
// ---------------------------------------------------------------------------
__global__ __launch_bounds__(256)
void ln_mod_kernel(const float* __restrict__ x, const float* __restrict__ mod,
                   const int* __restrict__ seq_ids, bf16* __restrict__ xm){
    __shared__ float rs[256], rq[256];
    const int l = blockIdx.x;
    const float* row = x + (size_t)l*1024;
    float s = 0.f, q = 0.f;
    for (int i = threadIdx.x; i < 1024; i += 256){ float v = row[i]; s += v; q += v*v; }
    rs[threadIdx.x] = s; rq[threadIdx.x] = q; __syncthreads();
    for (int o = 128; o > 0; o >>= 1){
        if (threadIdx.x < o){ rs[threadIdx.x] += rs[threadIdx.x+o]; rq[threadIdx.x] += rq[threadIdx.x+o]; }
        __syncthreads();
    }
    const float mu   = rs[0]*(1.f/1024.f);
    const float var  = rq[0]*(1.f/1024.f) - mu*mu;
    const float rstd = rsqrtf(var + 1e-6f);
    const int sid = seq_ids[l];
    const float* shiftp = mod + (size_t)sid*3072;
    const float* scalep = shiftp + 1024;
    for (int i = threadIdx.x; i < 1024; i += 256){
        float xn = (row[i]-mu)*rstd;
        xm[(size_t)l*1024 + i] = (bf16)((1.f + scalep[i])*xn + shiftp[i]);
    }
}

// ---------------------------------------------------------------------------
// Tiled bf16 WMMA GEMM:  C = A(MxK) @ B(NxK)^T
// Block tile 256x64, 8 waves (4x2), wave tile 64x32, K-tile 64 (2 wmma steps).
// Tiles staged via GLOBAL_LOAD_ASYNC_TO_LDS_B128 (ASYNCcnt) — no VGPR bounce.
// MODE 1: h = A@w1^T + b1 ; cols<3072 -> qkv (bf16), else gelu -> y[:,1024:]
// MODE 2: out = x + gate * (A@w2^T + b2)   (f32 out)
// ---------------------------------------------------------------------------
template<int MODE>
__global__ __launch_bounds__(256)
void gemm_bf16(const bf16* __restrict__ A, const bf16* __restrict__ B, int K,
               const float* __restrict__ bias,
               bf16* __restrict__ qkv_out, bf16* __restrict__ y_out,
               const float* __restrict__ x_res, const float* __restrict__ modv,
               const int* __restrict__ seq_ids, float* __restrict__ out){
    __shared__ __align__(16) bf16 As[256*72];   // 36 KB
    __shared__ __align__(16) bf16 Bs[64*72];    //  9 KB
    const int tid   = threadIdx.x;
    const int lane  = tid & 31;
    const int wave  = tid >> 5;
    const int wm    = wave >> 1;     // 0..3   (64-row wave tile)
    const int wn    = wave & 1;      // 0..1   (32-col wave tile)
    const int khalf = lane >> 4;
    const int lr    = lane & 15;
    const size_t m0 = (size_t)blockIdx.x * 256;
    const size_t n0 = (size_t)blockIdx.y * 64;

    v8f acc[4][2];
    #pragma unroll
    for (int i=0;i<4;++i)
        #pragma unroll
        for (int j=0;j<2;++j)
            #pragma unroll
            for (int r=0;r<8;++r) acc[i][j][r] = 0.f;

    const int arow = tid >> 3, akk = (tid & 7)*8;   // this thread's first A unit
    for (int k0 = 0; k0 < K; k0 += 64){
        // prefetch next K-tile while staging this one
        if (k0 + 64 < K){
            __builtin_prefetch(A + (m0+arow)*(size_t)K + k0 + 64 + akk, 0, 1);
            __builtin_prefetch(B + (n0+(arow & 63))*(size_t)K + k0 + 64 + akk, 0, 1);
        }
        // async-stage A (2048 16B units / 256 threads = 8 each) and B (2 each)
        #pragma unroll
        for (int i=0;i<8;++i){
            int u = tid + i*256;
            int row = u >> 3, kk = (u & 7)*8;
            async_g2l_b128((unsigned)(size_t)(&As[row*72 + kk]),
                           A + (m0+row)*(size_t)K + k0 + kk);
        }
        #pragma unroll
        for (int i=0;i<2;++i){
            int u = tid + i*256;
            int row = u >> 3, kk = (u & 7)*8;
            async_g2l_b128((unsigned)(size_t)(&Bs[row*72 + kk]),
                           B + (n0+row)*(size_t)K + k0 + kk);
        }
        wait_async0();
        __syncthreads();
        #pragma unroll
        for (int s=0;s<2;++s){
            v16bf af[4], bfr[2];
            #pragma unroll
            for (int t=0;t<4;++t){
                const bf16* base = &As[(wm*64 + t*16 + lr)*72 + s*32];
                af[t] = combine16(*(const v8bf*)(base + khalf*8),
                                  *(const v8bf*)(base + 16 + khalf*8));
            }
            #pragma unroll
            for (int t=0;t<2;++t){
                const bf16* base = &Bs[(wn*32 + t*16 + lr)*72 + s*32];
                bfr[t] = combine16(*(const v8bf*)(base + khalf*8), *(const v8bf*)(base + 16 + khalf*8));
            }
            #pragma unroll
            for (int i=0;i<4;++i)
                #pragma unroll
                for (int j=0;j<2;++j)
                    acc[i][j] = __builtin_amdgcn_wmma_f32_16x16x32_bf16(
                        false, af[i], false, bfr[j], (short)0, acc[i][j], false, false);
        }
        __syncthreads();
    }
    // epilogue
    #pragma unroll
    for (int i=0;i<4;++i){
        #pragma unroll
        for (int j=0;j<2;++j){
            #pragma unroll
            for (int r=0;r<8;++r){
                size_t m = m0 + wm*64 + i*16 + khalf*8 + r;
                int    n = (int)n0 + wn*32 + j*16 + lr;
                float  v = acc[i][j][r] + bias[n];
                if (MODE == 1){
                    if (n < 3072){
                        qkv_out[m*3072 + n] = (bf16)v;
                    } else {
                        float g = 0.5f*v*(1.0f + tanhf(0.7978845608028654f*(v + 0.044715f*v*v*v)));
                        y_out[m*5120 + 1024 + (n - 3072)] = (bf16)g;
                    }
                } else {
                    int sid = seq_ids[m];
                    out[m*1024 + n] = x_res[m*1024 + n] + modv[(size_t)sid*3072 + 2048 + n]*v;
                }
            }
        }
    }
}

// ---------------------------------------------------------------------------
// per-(l,head) RMSNorm + RoPE; qkv bf16 -> q_rope / k_rope bf16
// ---------------------------------------------------------------------------
__global__ __launch_bounds__(256)
void rope_kernel(const bf16* __restrict__ qkv, const float* __restrict__ pe,
                 const float* __restrict__ qsc, const float* __restrict__ ksc,
                 bf16* __restrict__ qout, bf16* __restrict__ kout){
    const int idx = blockIdx.x*256 + threadIdx.x;   // L*H threads
    const int l = idx >> 4, h = idx & 15;
    const bf16* qp = qkv + (size_t)l*3072 + h*64;
    const bf16* kp = qp + 1024;
    float sq = 0.f, sk = 0.f;
    for (int d = 0; d < 64; ++d){
        float a = (float)qp[d]; sq += a*a;
        float b = (float)kp[d]; sk += b*b;
    }
    const float rq = rsqrtf(sq*(1.f/64.f) + 1e-6f);
    const float rk = rsqrtf(sk*(1.f/64.f) + 1e-6f);
    const float* pl = pe + (size_t)l*128;
    bf16* qo = qout + ((size_t)l*16 + h)*64;
    bf16* ko = kout + ((size_t)l*16 + h)*64;
    for (int j = 0; j < 32; ++j){
        float a = pl[4*j+0], b = pl[4*j+1], c = pl[4*j+2], d2 = pl[4*j+3];
        float q0 = (float)qp[2*j]  *rq*qsc[2*j];
        float q1 = (float)qp[2*j+1]*rq*qsc[2*j+1];
        float k0 = (float)kp[2*j]  *rk*ksc[2*j];
        float k1 = (float)kp[2*j+1]*rk*ksc[2*j+1];
        qo[2*j]   = (bf16)(a*q0 + b*q1);
        qo[2*j+1] = (bf16)(c*q0 + d2*q1);
        ko[2*j]   = (bf16)(a*k0 + b*k1);
        ko[2*j+1] = (bf16)(c*k0 + d2*k1);
    }
}

// ---------------------------------------------------------------------------
// Flash attention: one wave per (16-query block, head). Ragged via seq table.
// S^T = K_blk @ Q^T (wmma), online softmax per lane-pair, O^T = V^T @ P^T.
// K fragments preloaded so loads pipeline under one wait; prefetch next block.
// ---------------------------------------------------------------------------
__global__ __launch_bounds__(128)
void attn_kernel(const bf16* __restrict__ qr, const bf16* __restrict__ kr,
                 const bf16* __restrict__ qkv, bf16* __restrict__ y){
    __shared__ __align__(16) bf16 sVT[4][64*40];
    __shared__ __align__(16) bf16 sP [4][16*40];
    const int wave  = threadIdx.x >> 5;
    const int lane  = threadIdx.x & 31;
    const int khalf = lane >> 4;
    const int lr    = lane & 15;
    const int head  = blockIdx.y*4 + wave;
    const int q0    = blockIdx.x * 16;
    const int starts[9] = {0,2048,3584,4608,6656,8448,9728,11776,13312};
    int s = 0;
    while (q0 >= starts[s+1]) ++s;
    const int kbeg = starts[s], kend = starts[s+1];
    bf16* myVT = sVT[wave];
    bf16* myP  = sP[wave];

    v16bf bq[2];
    {
        const bf16* qb = qr + ((size_t)(q0+lr)*16 + head)*64;
        #pragma unroll
        for (int t=0;t<2;++t){
            const bf16* p = qb + t*32;
            bq[t] = combine16(*(const v8bf*)(p + khalf*8), *(const v8bf*)(p + 16 + khalf*8));
        }
    }

    v8f oacc[4];
    #pragma unroll
    for (int dt=0;dt<4;++dt)
        #pragma unroll
        for (int r=0;r<8;++r) oacc[dt][r] = 0.f;
    float mi = -1e30f, li = 0.f;

    for (int kb = kbeg; kb < kend; kb += 32){
        if (kb + 32 < kend){
            __builtin_prefetch(kr + ((size_t)(kb + 32 + lr)*16 + head)*64, 0, 1);
            __builtin_prefetch(qkv + (size_t)(kb + 32 + lane)*3072 + 2048 + head*64, 0, 1);
        }
        // preload all K fragments for this key block (pipelined loads)
        v16bf ak[2][2];
        #pragma unroll
        for (int mt=0;mt<2;++mt){
            const bf16* kb_ = kr + ((size_t)(kb + mt*16 + lr)*16 + head)*64;
            #pragma unroll
            for (int ds=0;ds<2;++ds){
                const bf16* p = kb_ + ds*32;
                ak[mt][ds] = combine16(*(const v8bf*)(p + khalf*8),
                                       *(const v8bf*)(p + 16 + khalf*8));
            }
        }
        { // stage V^T: lane owns key kb+lane
            const bf16* vrow = qkv + (size_t)(kb+lane)*3072 + 2048 + head*64;
            #pragma unroll
            for (int d0=0; d0<64; d0+=8){
                v8bf c = *(const v8bf*)(vrow + d0);
                #pragma unroll
                for (int e=0;e<8;++e) myVT[(d0+e)*40 + lane] = c[e];
            }
        }
        v8f sacc[2];
        #pragma unroll
        for (int mt=0;mt<2;++mt)
            #pragma unroll
            for (int r=0;r<8;++r) sacc[mt][r] = 0.f;
        #pragma unroll
        for (int mt=0;mt<2;++mt)
            #pragma unroll
            for (int ds=0;ds<2;++ds)
                sacc[mt] = __builtin_amdgcn_wmma_f32_16x16x32_bf16(
                    false, ak[mt][ds], false, bq[ds], (short)0, sacc[mt], false, false);
        // online softmax over 32 keys; query = lr, partner lane = lane^16
        float sv[16];
        float mb = -3.0e30f;
        #pragma unroll
        for (int mt=0;mt<2;++mt)
            #pragma unroll
            for (int r=0;r<8;++r){
                float t = sacc[mt][r]*0.125f;
                sv[mt*8+r] = t;
                mb = fmaxf(mb, t);
            }
        mb = fmaxf(mb, __shfl_xor(mb, 16, 32));
        float mnew = fmaxf(mi, mb);
        float corr = __expf(mi - mnew);
        float ls = 0.f;
        #pragma unroll
        for (int mt=0;mt<2;++mt)
            #pragma unroll
            for (int r=0;r<8;++r){
                float p = __expf(sv[mt*8+r] - mnew);
                ls += p;
                myP[lr*40 + mt*16 + khalf*8 + r] = (bf16)p;
            }
        ls += __shfl_xor(ls, 16, 32);
        li = li*corr + ls;
        mi = mnew;
        #pragma unroll
        for (int dt=0;dt<4;++dt)
            #pragma unroll
            for (int r=0;r<8;++r) oacc[dt][r] *= corr;
        v16bf bp;
        {
            const bf16* p = myP + lr*40;
            bp = combine16(*(const v8bf*)(p + khalf*8), *(const v8bf*)(p + 16 + khalf*8));
        }
        #pragma unroll
        for (int dt=0;dt<4;++dt){
            const bf16* vb = myVT + (dt*16 + lr)*40;
            v16bf av = combine16(*(const v8bf*)(vb + khalf*8),
                                 *(const v8bf*)(vb + 16 + khalf*8));
            oacc[dt] = __builtin_amdgcn_wmma_f32_16x16x32_bf16(
                false, av, false, bp, (short)0, oacc[dt], false, false);
        }
    }
    const float rinv = 1.f / li;
    #pragma unroll
    for (int dt=0;dt<4;++dt)
        #pragma unroll
        for (int r=0;r<8;++r){
            int d = dt*16 + khalf*8 + r;
            y[(size_t)(q0+lr)*5120 + head*64 + d] = (bf16)(oacc[dt][r]*rinv);
        }
}

// ---------------------------------------------------------------------------
extern "C" void kernel_launch(void* const* d_in, const int* in_sizes, int n_in,
                              void* d_out, int out_size, void* d_ws, size_t ws_size,
                              hipStream_t stream){
    const float* x   = (const float*)d_in[0];
    const float* vec = (const float*)d_in[1];
    const float* pe  = (const float*)d_in[2];
    const int*   sid = (const int*)  d_in[3];
    const float* mw  = (const float*)d_in[4];
    const float* mbv = (const float*)d_in[5];
    const float* w1  = (const float*)d_in[6];
    const float* b1  = (const float*)d_in[7];
    const float* w2  = (const float*)d_in[8];
    const float* b2  = (const float*)d_in[9];
    const float* qs  = (const float*)d_in[10];
    const float* ks  = (const float*)d_in[11];
    float* out = (float*)d_out;

    char* ws = (char*)d_ws;
    float* mod  = (float*)(ws);                    // 8*3072*4        =    98,304
    bf16* xm    = (bf16*)(ws + 98304);             // 13312*1024*2    = 27,262,976
    bf16* w1b   = (bf16*)(ws + 27361280);          // 7168*1024*2     = 14,680,064
    bf16* w2b   = (bf16*)(ws + 42041344);          // 1024*5120*2     = 10,485,760
    bf16* qkvb  = (bf16*)(ws + 52527104);          // 13312*3072*2    = 81,788,928
    bf16* qrope = (bf16*)(ws + 134316032);         // 13312*1024*2    = 27,262,976
    bf16* krope = (bf16*)(ws + 161579008);         // 13312*1024*2    = 27,262,976
    bf16* yb    = (bf16*)(ws + 188841984);         // 13312*5120*2    = 136,314,880

    f2bf_kernel<<<7168, 256, 0, stream>>>(w1, w1b, (size_t)7168*1024);
    f2bf_kernel<<<5120, 256, 0, stream>>>(w2, w2b, (size_t)1024*5120);
    mod_kernel<<<dim3(12,8), 256, 0, stream>>>(vec, mw, mbv, mod);
    ln_mod_kernel<<<13312, 256, 0, stream>>>(x, mod, sid, xm);
    gemm_bf16<1><<<dim3(52,112), 256, 0, stream>>>(xm, w1b, 1024, b1, qkvb, yb,
                                                   nullptr, nullptr, nullptr, nullptr);
    rope_kernel<<<832, 256, 0, stream>>>(qkvb, pe, qs, ks, qrope, krope);
    attn_kernel<<<dim3(832,4), 128, 0, stream>>>(qrope, krope, qkvb, yb);
    gemm_bf16<2><<<dim3(52,16), 256, 0, stream>>>(yb, w2b, 5120, b2,
                                                  nullptr, nullptr, x, mod, sid, out);
}
